// Head_10393820856499
// MI455X (gfx1250) — matrix operand order repaired and checked by reading, not compiled, not run on hardware
//
#include <hip/hip_runtime.h>

#define B_ 1024
#define H_ 512
#define N_ 16384
#define M_ 64

typedef _Float16 v16h __attribute__((ext_vector_type(16)));
typedef _Float16 h8_t __attribute__((ext_vector_type(8)));
typedef float    v8f  __attribute__((ext_vector_type(8)));

// ---------------- Kernel 1: controller projections -------------------------
__global__ __launch_bounds__(256) void proj_kernel(
    const float* __restrict__ h,
    const float* __restrict__ Wk, const float* __restrict__ bk,
    const float* __restrict__ Wb, const float* __restrict__ bb,
    const float* __restrict__ Wg, const float* __restrict__ bg,
    const float* __restrict__ Ws, const float* __restrict__ bs,
    const float* __restrict__ Wm, const float* __restrict__ bm,
    _Float16* __restrict__ k16, float* __restrict__ knorm,
    float* __restrict__ beta, float* __restrict__ gg,
    float* __restrict__ s3, float* __restrict__ gam)
{
    __shared__ float hs[H_];
    __shared__ float outs[70];
    __shared__ float kc2[64];
    const int b = blockIdx.x, t = threadIdx.x;
    hs[t]       = h[b * H_ + t];
    hs[t + 256] = h[b * H_ + t + 256];
    __syncthreads();
    if (t < 70) {
        const float* wrow; float bias;
        if (t < 64)       { wrow = Wk + t * H_;        bias = bk[t];      }
        else if (t == 64) { wrow = Wb;                 bias = bb[0];      }
        else if (t == 65) { wrow = Wg;                 bias = bg[0];      }
        else if (t < 69)  { wrow = Ws + (t - 66) * H_; bias = bs[t - 66]; }
        else              { wrow = Wm;                 bias = bm[0];      }
        float acc = bias;
        for (int j = 0; j < H_; ++j) acc += hs[j] * wrow[j];
        outs[t] = acc;
    }
    __syncthreads();
    if (t < 64) {
        float kc = fminf(fmaxf(outs[t], 0.f), 1.f);   // clip(.,0,1)
        k16[b * 64 + t] = (_Float16)kc;
        kc2[t] = kc * kc;
    }
    __syncthreads();
    if (t == 0) {
        float ss = 0.f;
        for (int j = 0; j < 64; ++j) ss += kc2[j];
        knorm[b] = sqrtf(ss);
        beta[b]  = fmaxf(outs[64], 0.f);                       // relu
        gg[b]    = fminf(fmaxf(outs[65], 0.f), 1.f);           // clip
        float x0 = outs[66], x1 = outs[67], x2 = outs[68];
        float mx = fmaxf(x0, fmaxf(x1, x2));
        float e0 = __expf(x0 - mx), e1 = __expf(x1 - mx), e2 = __expf(x2 - mx);
        float inv = __builtin_amdgcn_rcpf(e0 + e1 + e2);
        s3[b * 3 + 0] = e0 * inv; s3[b * 3 + 1] = e1 * inv; s3[b * 3 + 2] = e2 * inv;
        gam[b] = 1.f + fmaxf(outs[69], 0.f);
    }
}

// ---------------- Kernel 2: m_t -> f16 + row norms -------------------------
__global__ __launch_bounds__(256) void mprep_kernel(
    const float* __restrict__ m, _Float16* __restrict__ m16,
    float* __restrict__ mnorm)
{
    const int n = blockIdx.x * 256 + threadIdx.x;
    const float* row = m + (size_t)n * 64;
    float s = 0.f;
    #pragma unroll 8
    for (int j = 0; j < 64; ++j) {
        float v = row[j];
        s += v * v;
        m16[(size_t)n * 64 + j] = (_Float16)v;
    }
    mnorm[n] = sqrtf(s);
}

// ---------------- Kernel 3: WMMA GEMM  scores = scale * (k @ m^T) ----------
// Each wave: one B fragment (16 cols, K=64) reused across 4 row tiles.
// 8 x v_wmma_f32_16x16x32_f16 per wave, fast-rcp epilogue.
__global__ __launch_bounds__(256) void gemm_kernel(
    const _Float16* __restrict__ k16, const _Float16* __restrict__ m16,
    const float* __restrict__ knorm, const float* __restrict__ mnorm,
    const float* __restrict__ beta, float* __restrict__ scores)
{
    const int wave = threadIdx.x >> 5;
    const int lane = threadIdx.x & 31;
    const int rg       = blockIdx.x;                       // row group: 4 tiles
    const int col_base = (blockIdx.y * 8 + wave) << 4;
    const int mi = lane & 15;          // M (for A) / N (for B, C)
    const int kh = lane >> 4;          // which K-half this lane holds

    // B fragment: lane holds column (col_base+mi); 16 consecutive K values
    // starting at 16*kh -> one contiguous 32B chunk of the m16 row.
    const _Float16* brow = m16 + (size_t)(col_base + mi) * 64;
    const v16h b0 = *(const v16h*)(brow +  0 + 16 * kh);
    const v16h b1 = *(const v16h*)(brow + 32 + 16 * kh);
    const float mn = mnorm[col_base + mi];

    #pragma unroll
    for (int rt = 0; rt < 4; ++rt) {
        const int row_base = (rg << 6) + (rt << 4);
        const _Float16* arow = k16 + (size_t)(row_base + mi) * 64;
        // A fragment layout (16-bit A 16x32): lanes hold row mi; elements 0-15 = K
        // [16*kh, 16*kh+16), matching B. Contiguous 8-half chunks.
        h8_t a0lo = *(const h8_t*)(arow +  0 + 16 * kh);
        h8_t a0hi = *(const h8_t*)(arow +  8 + 16 * kh);
        h8_t a1lo = *(const h8_t*)(arow + 32 + 16 * kh);
        h8_t a1hi = *(const h8_t*)(arow + 40 + 16 * kh);
        v16h a0, a1;
        #pragma unroll
        for (int i = 0; i < 8; ++i) {
            a0[i] = a0lo[i]; a0[i + 8] = a0hi[i];
            a1[i] = a1lo[i]; a1[i + 8] = a1hi[i];
        }
        v8f c = {};
        c = __builtin_amdgcn_wmma_f32_16x16x32_f16(false, a0, false, b0, (short)0, c, false, false);
        c = __builtin_amdgcn_wmma_f32_16x16x32_f16(false, a1, false, b1, (short)0, c, false, false);

        // Epilogue: score = num * beta_r * rcp(||k_r||*||m_n|| + eps)
        #pragma unroll
        for (int i = 0; i < 8; ++i) {
            int r = row_base + i + 8 * kh;             // C layout: M = i + 8*kh
            float scale = beta[r] * __builtin_amdgcn_rcpf(knorm[r] * mn + 1e-6f);
            scores[(size_t)r * N_ + col_base + mi] = c[i] * scale;
        }
    }
}

// ---------------- Kernel 4: softmax stats per row --------------------------
__global__ __launch_bounds__(256) void stats_kernel(
    const float* __restrict__ sc, float* __restrict__ rmax,
    float* __restrict__ rinv, float* __restrict__ wsum)
{
    __shared__ float red[256];
    const int b = blockIdx.x, t = threadIdx.x;
    const float* row = sc + (size_t)b * N_;
    float mx = -3.4e38f;
    for (int j = t; j < N_; j += 256) mx = fmaxf(mx, row[j]);
    red[t] = mx; __syncthreads();
    for (int s = 128; s > 0; s >>= 1) { if (t < s) red[t] = fmaxf(red[t], red[t + s]); __syncthreads(); }
    const float rm = red[0];
    __syncthreads();
    float sm = 0.f;
    for (int j = t; j < N_; j += 256) sm += __expf(row[j] - rm);
    red[t] = sm; __syncthreads();
    for (int s = 128; s > 0; s >>= 1) { if (t < s) red[t] += red[t + s]; __syncthreads(); }
    if (t == 0) { rmax[b] = rm; rinv[b] = 1.f / red[0]; wsum[b] = 0.f; }
}

// ---------------- Kernel 5: gate + shift-conv + sharpen + partial sums -----
__global__ __launch_bounds__(256) void ew_kernel(
    const float* __restrict__ sc, const float* __restrict__ w_tm1,
    const float* __restrict__ rmax, const float* __restrict__ rinv,
    const float* __restrict__ gg, const float* __restrict__ s3,
    const float* __restrict__ gam, float* __restrict__ out,
    float* __restrict__ wsum)
{
    __shared__ float wg[258];
    __shared__ float red[256];
    const int b = blockIdx.y, t = threadIdx.x;
    const int n0 = blockIdx.x * 256;
    const int n  = n0 + t;
    const float rm = rmax[b], ri = rinv[b], gb = gg[b];
    const float* srow = sc    + (size_t)b * N_;
    const float* wrow = w_tm1 + (size_t)b * N_;

    wg[t + 1] = gb * (__expf(srow[n] - rm) * ri) + (1.f - gb) * wrow[n];
    if (t == 0)
        wg[0]   = (n0 > 0)        ? gb * (__expf(srow[n0 - 1]   - rm) * ri) + (1.f - gb) * wrow[n0 - 1]   : 0.f;
    if (t == 255)
        wg[257] = (n0 + 256 < N_) ? gb * (__expf(srow[n0 + 256] - rm) * ri) + (1.f - gb) * wrow[n0 + 256] : 0.f;
    __syncthreads();

    // w_tilde[n] = s0*wg[n-1] + s1*wg[n] + s2*wg[n+1]  (zero pad at row edges)
    float wt = s3[b * 3 + 0] * wg[t] + s3[b * 3 + 1] * wg[t + 1] + s3[b * 3 + 2] * wg[t + 2];
    float w  = __expf(gam[b] * __logf(wt + 1e-6f));   // (wt+1e-6)^gamma, wt>=0
    out[(size_t)b * N_ + n] = w;

    red[t] = w; __syncthreads();
    for (int s = 128; s > 0; s >>= 1) { if (t < s) red[t] += red[t + s]; __syncthreads(); }
    if (t == 0) atomicAdd(&wsum[b], red[0]);
}

// ---------------- Kernel 6: final normalization ----------------------------
__global__ __launch_bounds__(256) void norm_kernel(
    float* __restrict__ out, const float* __restrict__ wsum)
{
    const size_t idx = (size_t)blockIdx.x * 256 + threadIdx.x;
    const int b = (int)(idx >> 14);                 // N = 16384 = 2^14
    out[idx] = out[idx] * __builtin_amdgcn_rcpf(wsum[b]);
}

// ---------------------------------------------------------------------------
extern "C" void kernel_launch(void* const* d_in, const int* in_sizes, int n_in,
                              void* d_out, int out_size, void* d_ws, size_t ws_size,
                              hipStream_t stream)
{
    (void)in_sizes; (void)n_in; (void)out_size; (void)ws_size;
    const float* h_t   = (const float*)d_in[0];
    const float* w_tm1 = (const float*)d_in[1];
    const float* m_t   = (const float*)d_in[2];
    const float* Wk    = (const float*)d_in[3];
    const float* bk    = (const float*)d_in[4];
    const float* Wb    = (const float*)d_in[5];
    const float* bb    = (const float*)d_in[6];
    const float* Wg    = (const float*)d_in[7];
    const float* bg    = (const float*)d_in[8];
    const float* Ws    = (const float*)d_in[9];
    const float* bs    = (const float*)d_in[10];
    const float* Wm    = (const float*)d_in[11];
    const float* bm    = (const float*)d_in[12];
    float* out = (float*)d_out;

    char* ws = (char*)d_ws;
    float*    scores = (float*)    (ws);               // 64 MB
    _Float16* k16    = (_Float16*) (ws + 67108864);    // 128 KB
    _Float16* m16    = (_Float16*) (ws + 67239936);    // 2 MB
    float*    knorm  = (float*)    (ws + 69337088);    // 4 KB
    float*    mnorm  = (float*)    (ws + 69341184);    // 64 KB
    float*    beta   = (float*)    (ws + 69406720);    // 4 KB
    float*    gg     = (float*)    (ws + 69410816);    // 4 KB
    float*    gam    = (float*)    (ws + 69414912);    // 4 KB
    float*    s3     = (float*)    (ws + 69419008);    // 12 KB
    float*    rmax   = (float*)    (ws + 69431296);    // 4 KB
    float*    rinv   = (float*)    (ws + 69435392);    // 4 KB
    float*    wsum   = (float*)    (ws + 69439488);    // 4 KB

    proj_kernel<<<B_, 256, 0, stream>>>(h_t, Wk, bk, Wb, bb, Wg, bg, Ws, bs,
                                        Wm, bm, k16, knorm, beta, gg, s3, gam);
    mprep_kernel<<<N_ / 256, 256, 0, stream>>>(m_t, m16, mnorm);
    gemm_kernel<<<dim3(B_ / 64, N_ / 16 / 8), 256, 0, stream>>>(
        k16, m16, knorm, mnorm, beta, scores);
    stats_kernel<<<B_, 256, 0, stream>>>(scores, rmax, rinv, wsum);
    ew_kernel<<<dim3(N_ / 256, B_), 256, 0, stream>>>(
        scores, w_tm1, rmax, rinv, gg, s3, gam, out, wsum);
    norm_kernel<<<(B_ * N_) / 256, 256, 0, stream>>>(out, wsum);
}
